// RNNModel_49082886258848
// MI455X (gfx1250) — matrix operand order repaired and checked by reading, not compiled, not run on hardware
//
#include <hip/hip_runtime.h>
#include <math.h>

// ---------------------------------------------------------------------------
// DARTS-RNN forward for MI455X (gfx1250), wave32 WMMA via
// v_wmma_f32_16x16x32_f16. fp32 fidelity kept with a 2-term f16 hi/lo split:
// A*B ~= Ah*Bh + Ah*Bl + Al*Bh (3 WMMAs per K-chunk, fp32 accumulate).
// Weights are pre-transposed column-major so B fragments are single 32B loads.
// Independent genotype steps are batched into one launch via blockIdx.y.
// ---------------------------------------------------------------------------

typedef __attribute__((ext_vector_type(16))) _Float16 v16h;
typedef __attribute__((ext_vector_type(8)))  _Float16 v8h;
typedef __attribute__((ext_vector_type(8)))  float    v8f;

#define B_   64
#define T_   128
#define D_   850
#define D2_  1700
#define NC_  42
#define KC   864     // 850 padded to multiple of 32 (pad is zero everywhere)
#define KW0  1728    // two 864 K-segments for [x_t | h_prev] @ W0
#define KAG  928     // 910 padded to multiple of 32
#define BD   (B_ * D_)

// GENOTYPE tables packed as immediates: preds[i] (3 bits), acts[i] (2 bits)
// preds = {0,1,1,1,2,5,3,5}; acts = {sig,relu,relu,id,tanh,sig,tanh,relu}
//                                 = {2,1,1,3,0,2,0,1} (0=tanh 1=relu 2=sig 3=id)
#define PREDS_PACKED (0 | (1<<3) | (1<<6) | (1<<9) | (2<<12) | (5<<15) | (3<<18) | (5<<21))
#define ACTS_PACKED  (2 | (1<<2) | (1<<4) | (3<<6) | (0<<8) | (2<<10) | (0<<12) | (1<<14))

#define WMMA(a, b, c) \
  __builtin_amdgcn_wmma_f32_16x16x32_f16(false, (a), false, (b), (short)0, (c), false, false)

__device__ __forceinline__ v8f v8f_zero() {
  v8f z = {0.f, 0.f, 0.f, 0.f, 0.f, 0.f, 0.f, 0.f};
  return z;
}

// A fragment (16x32 f16): lane holds K [16*hi, 16*hi+16)
__device__ __forceinline__ v16h ld_afrag(const _Float16* p) {
  union { v16h v; v8h h[2]; } u;
  u.h[0] = *(const v8h*)(p);
  u.h[1] = *(const v8h*)(p + 8);
  return u.v;
}

__device__ __forceinline__ void splitf(float x, _Float16* h, _Float16* l) {
  _Float16 hh = (_Float16)x;
  *h = hh;
  *l = (_Float16)(x - (float)hh);
}

// ---- generic split-GEMM mainloop: one 16x16 tile, B stored col-major [n][k]
__device__ __forceinline__ v8f split_mm(const _Float16* __restrict__ Ah,
                                        const _Float16* __restrict__ Al, int lda, int m0,
                                        const _Float16* __restrict__ Bh,
                                        const _Float16* __restrict__ Bl, int ldb,
                                        int nc, int Kpad, v8f acc)
{
  const int lane = threadIdx.x & 31;
  const int lo = lane & 15, hi = lane >> 4;
  const _Float16* aph = Ah + (size_t)(m0 + lo) * lda + 16 * hi;
  const _Float16* apl = Al + (size_t)(m0 + lo) * lda + 16 * hi;
  const _Float16* bph = Bh + (size_t)nc * ldb + 16 * hi;
  const _Float16* bpl = Bl + (size_t)nc * ldb + 16 * hi;
  for (int kk = 0; kk < Kpad; kk += 32) {
    v16h a0 = ld_afrag(aph + kk);
    v16h a1 = ld_afrag(apl + kk);
    v16h b0 = *(const v16h*)(bph + kk);
    v16h b1 = *(const v16h*)(bpl + kk);
    acc = WMMA(a0, b0, acc);
    acc = WMMA(a0, b1, acc);
    acc = WMMA(a1, b0, acc);
  }
  return acc;
}

// ---- cell mainloop: shares A fragments between the c-tile and h-tile
struct CH { v8f c; v8f h; };

__device__ __forceinline__ CH cell_seg(const _Float16* __restrict__ Ah,
                                       const _Float16* __restrict__ Al, int m0,
                                       const _Float16* __restrict__ Bh,
                                       const _Float16* __restrict__ Bl, int ldb,
                                       int ncc, int nch, CH acc)
{
  const int lane = threadIdx.x & 31;
  const int lo = lane & 15, hi = lane >> 4;
  const _Float16* aph = Ah + (size_t)(m0 + lo) * KC + 16 * hi;
  const _Float16* apl = Al + (size_t)(m0 + lo) * KC + 16 * hi;
  const _Float16* bch = Bh + (size_t)ncc * ldb + 16 * hi;
  const _Float16* bcl = Bl + (size_t)ncc * ldb + 16 * hi;
  const _Float16* bhh = Bh + (size_t)nch * ldb + 16 * hi;
  const _Float16* bhl = Bl + (size_t)nch * ldb + 16 * hi;
  for (int kk = 0; kk < KC; kk += 32) {
    v16h a0 = ld_afrag(aph + kk);
    v16h a1 = ld_afrag(apl + kk);
    v16h b0 = *(const v16h*)(bch + kk);
    v16h b1 = *(const v16h*)(bcl + kk);
    v16h b2 = *(const v16h*)(bhh + kk);
    v16h b3 = *(const v16h*)(bhl + kk);
    acc.c = WMMA(a0, b0, acc.c);
    acc.c = WMMA(a0, b1, acc.c);
    acc.c = WMMA(a1, b0, acc.c);
    acc.h = WMMA(a0, b2, acc.h);
    acc.h = WMMA(a0, b3, acc.h);
    acc.h = WMMA(a1, b2, acc.h);
  }
  return acc;
}

// ---- shared gate epilogue: S[m,n] = sp + sigmoid(c)*(act(h) - sp)
__device__ __forceinline__ void gate_epilogue(CH acc, int m0, int n0,
                                              const float* SP, float* S,
                                              _Float16* Sh, _Float16* Sl, int act)
{
  const int lane = threadIdx.x & 31;
  const int lo = lane & 15, hi = lane >> 4;
  const int n  = n0 + lo;
  const int mb = m0 + hi * 8;
  if (n < D_) {
#pragma unroll
    for (int r = 0; r < 8; ++r) {
      int m = mb + r;
      float sp = SP[m * D_ + n];
      float g  = 1.f / (1.f + expf(-acc.c[r]));
      float hv = acc.h[r];
      float a;
      if (act == 0)      a = tanhf(hv);
      else if (act == 1) a = fmaxf(hv, 0.f);
      else if (act == 2) a = 1.f / (1.f + expf(-hv));
      else               a = hv;
      float s = sp + g * (a - sp);
      S[m * D_ + n] = s;
      splitf(s, &Sh[m * KC + n], &Sl[m * KC + n]);
    }
  } else {                                // zero the K-pad columns [850,864)
#pragma unroll
    for (int r = 0; r < 8; ++r) {
      int m = mb + r;
      Sh[m * KC + n] = (_Float16)0.f;
      Sl[m * KC + n] = (_Float16)0.f;
    }
  }
}

// ----------------- GEMM0: s0 from [x_t | h_prev] @ W0 (tanh gate) ----------
__global__ void __launch_bounds__(128)
gemm0(const _Float16* __restrict__ Xth, const _Float16* __restrict__ Xtl,
      const _Float16* __restrict__ hpH, const _Float16* __restrict__ hpL,
      const _Float16* __restrict__ W0Th, const _Float16* __restrict__ W0Tl,
      const float* __restrict__ hprev,
      float* __restrict__ S, _Float16* __restrict__ Sh, _Float16* __restrict__ Sl)
{
  const int wave = threadIdx.x >> 5;     // wave32: 4 waves cover M=64
  const int m0   = wave * 16;
  const int n0   = blockIdx.x * 16;      // 54 N-tiles cover 850
  const int lane = threadIdx.x & 31;
  const int lo = lane & 15;
  int ncc = n0 + lo;
  int nch = ncc + D_;
  if (nch >= D2_) nch = 0;
  CH acc; acc.c = v8f_zero(); acc.h = v8f_zero();
  acc = cell_seg(Xth, Xtl, m0, W0Th,      W0Tl,      KW0, ncc, nch, acc);
  acc = cell_seg(hpH, hpL, m0, W0Th + KC, W0Tl + KC, KW0, ncc, nch, acc);
  gate_epilogue(acc, m0, n0, hprev, S, Sh, Sl, /*tanh*/0);
}

// --------- grouped genotype-step GEMMs: blockIdx.y picks step id -----------
// steps: packed step indices, 8 bits each. Step i reads state preds[i],
// weights WsT[i], writes state i+1.
__global__ void __launch_bounds__(128)
step_gemm(_Float16* stHb, _Float16* stLb, float* Sstb,
          const _Float16* __restrict__ WsTh, const _Float16* __restrict__ WsTl,
          int steps)
{
  const int s   = (steps >> (8 * blockIdx.y)) & 0xff;
  const int p   = (PREDS_PACKED >> (3 * s)) & 7;
  const int act = (ACTS_PACKED  >> (2 * s)) & 3;
  const _Float16* A1h = stHb + (size_t)p * B_ * KC;
  const _Float16* A1l = stLb + (size_t)p * B_ * KC;
  const float*    SP  = Sstb + (size_t)p * BD;
  const _Float16* Bh  = WsTh + (size_t)s * D2_ * KC;
  const _Float16* Bl  = WsTl + (size_t)s * D2_ * KC;
  float*    S  = Sstb + (size_t)(s + 1) * BD;
  _Float16* Sh = stHb + (size_t)(s + 1) * B_ * KC;
  _Float16* Sl = stLb + (size_t)(s + 1) * B_ * KC;

  const int wave = threadIdx.x >> 5;
  const int m0   = wave * 16;
  const int n0   = blockIdx.x * 16;
  const int lane = threadIdx.x & 31;
  const int lo = lane & 15;
  int ncc = n0 + lo;
  int nch = ncc + D_;
  if (nch >= D2_) nch = 0;
  CH acc; acc.c = v8f_zero(); acc.h = v8f_zero();
  acc = cell_seg(A1h, A1l, m0, Bh, Bl, KC, ncc, nch, acc);
  gate_epilogue(acc, m0, n0, SP, S, Sh, Sl, act);
}

// ------------------------- aggregation GEMM --------------------------------
__global__ void __launch_bounds__(128)
agg_gemm(const _Float16* __restrict__ Ch, const _Float16* __restrict__ Cl,
         const _Float16* __restrict__ Wh, const _Float16* __restrict__ Wl,
         const float* __restrict__ aggB,
         _Float16* __restrict__ Xh, _Float16* __restrict__ Xl) // [T][B][864]
{
  const int wave = threadIdx.x >> 5;
  const int m0   = (blockIdx.y * 4 + wave) * 16;   // 8192 rows
  const int n0   = blockIdx.x * 16;
  const int lane = threadIdx.x & 31;
  const int lo = lane & 15, hi = lane >> 4;
  int nc = n0 + lo;
  if (nc >= D_) nc = 0;
  v8f acc = split_mm(Ch, Cl, KAG, m0, Wh, Wl, KAG, nc, KAG, v8f_zero());
  const int n = n0 + lo;
#pragma unroll
  for (int r = 0; r < 8; ++r) {
    int row = m0 + hi * 8 + r;            // row = b*T + t
    int b = row >> 7, t = row & 127;
    size_t o = ((size_t)t * B_ + b) * KC + n;
    if (n < D_) {
      float v = acc[r] + aggB[n];
      splitf(v, &Xh[o], &Xl[o]);
    } else {
      Xh[o] = (_Float16)0.f; Xl[o] = (_Float16)0.f;
    }
  }
}

// ---------------------------- decoder GEMM ---------------------------------
__global__ void __launch_bounds__(128)
dec_gemm(const _Float16* __restrict__ Ah, const _Float16* __restrict__ Al,
         const _Float16* __restrict__ Wh, const _Float16* __restrict__ Wl,
         const float* __restrict__ decB, float* __restrict__ logit)
{
  const int wave = threadIdx.x >> 5;
  const int m0   = wave * 16;
  const int n0   = blockIdx.x * 16;        // 3 tiles cover 42
  const int lane = threadIdx.x & 31;
  const int lo = lane & 15, hi = lane >> 4;
  int nc = n0 + lo;
  if (nc >= NC_) nc = 0;
  v8f acc = split_mm(Ah, Al, KC, m0, Wh, Wl, KC, nc, KC, v8f_zero());
  const int n = n0 + lo;
  if (n < NC_) {
    float bias = decB[n];
#pragma unroll
    for (int r = 0; r < 8; ++r)
      logit[(m0 + hi * 8 + r) * NC_ + n] = acc[r] * (1.f / (float)T_) + bias;
  }
}

// --------------------------- data prep kernels -----------------------------
__global__ void zero_f32(float* p, int n) {
  int i = blockIdx.x * 256 + threadIdx.x;
  if (i < n) p[i] = 0.f;
}

__global__ void build_combined(const int* __restrict__ tokens,
                               const int* __restrict__ ner,
                               const int* __restrict__ pos,
                               const float* __restrict__ encW,
                               const float* __restrict__ nerW,
                               const float* __restrict__ posW,
                               _Float16* __restrict__ Ch, _Float16* __restrict__ Cl)
{
  int i = blockIdx.x * 256 + threadIdx.x;
  if (i >= 8192 * KAG) return;
  int r = i / KAG, c = i % KAG;
  float v = 0.f;
  if (c < 850)      v = encW[(size_t)tokens[r] * 850 + c];
  else if (c < 880) v = nerW[ner[r] * 30 + (c - 850)];
  else if (c < 910) v = posW[pos[r] * 30 + (c - 880)];
  splitf(v, &Ch[i], &Cl[i]);
}

// W0 [1700][1700] -> col-major [n][k'] with k' = [850 x | 14 pad | 850 h | 14 pad]
__global__ void conv_W0T(const float* __restrict__ W0,
                         _Float16* __restrict__ Th, _Float16* __restrict__ Tl)
{
  int i = blockIdx.x * 256 + threadIdx.x;
  if (i >= D2_ * KW0) return;
  int n = i / KW0, kp = i % KW0;
  float v = 0.f;
  if (kp < D_)                       v = W0[(size_t)kp * D2_ + n];
  else if (kp >= KC && kp < KC + D_) v = W0[(size_t)(kp - (KC - D_)) * D2_ + n];
  splitf(v, &Th[i], &Tl[i]);
}

// Ws [8][850][1700] -> 8 x col-major [1700][864]
__global__ void conv_WsT(const float* __restrict__ Ws,
                         _Float16* __restrict__ Th, _Float16* __restrict__ Tl)
{
  int i = blockIdx.x * 256 + threadIdx.x;
  if (i >= 8 * D2_ * KC) return;
  int s = i / (D2_ * KC);
  int r = i % (D2_ * KC);
  int n = r / KC, k = r % KC;
  float v = (k < D_) ? Ws[((size_t)s * D_ + k) * D2_ + n] : 0.f;
  splitf(v, &Th[i], &Tl[i]);
}

// agg_W [850][910] is already [N][K] row-major -> pad K to 928
__global__ void conv_aggT(const float* __restrict__ aggW,
                          _Float16* __restrict__ Th, _Float16* __restrict__ Tl)
{
  int i = blockIdx.x * 256 + threadIdx.x;
  if (i >= D_ * KAG) return;
  int n = i / KAG, k = i % KAG;
  float v = (k < 910) ? aggW[(size_t)n * 910 + k] : 0.f;
  splitf(v, &Th[i], &Tl[i]);
}

// dec_W [42][850] -> pad K to 864
__global__ void conv_decT(const float* __restrict__ decW,
                          _Float16* __restrict__ Th, _Float16* __restrict__ Tl)
{
  int i = blockIdx.x * 256 + threadIdx.x;
  if (i >= NC_ * KC) return;
  int n = i / KC, k = i % KC;
  float v = (k < D_) ? decW[(size_t)n * D_ + k] : 0.f;
  splitf(v, &Th[i], &Tl[i]);
}

// fp32 [64][850] -> f16 hi/lo [64][864] (pad zero)
__global__ void conv_pair64(const float* __restrict__ src,
                            _Float16* __restrict__ Th, _Float16* __restrict__ Tl)
{
  int i = blockIdx.x * 256 + threadIdx.x;
  if (i >= B_ * KC) return;
  int b = i / KC, n = i % KC;
  float v = (n < D_) ? src[b * D_ + n] : 0.f;
  splitf(v, &Th[i], &Tl[i]);
}

// mean of states[1..8] -> h_prev (fp32 + f16 pair), masked accumulate, raw[-1]
__global__ void cell_reduce(const float* __restrict__ states,
                            float* __restrict__ hprev,
                            _Float16* __restrict__ hpH, _Float16* __restrict__ hpL,
                            float* __restrict__ accum,
                            const float* __restrict__ masks,
                            float* __restrict__ nh, int t)
{
  int i = blockIdx.x * 256 + threadIdx.x;    // over 64*864
  if (i >= B_ * KC) return;
  int b = i / KC, n = i % KC;
  if (n < D_) {
    int idx = b * D_ + n;
    float s = 0.f;
#pragma unroll
    for (int j = 1; j <= 8; ++j) s += states[j * BD + idx];   // CONCAT = 1..8
    s *= 0.125f;
    hprev[idx] = s;
    splitf(s, &hpH[i], &hpL[i]);
    float raw = s * masks[b * T_ + t];
    accum[idx] += raw;
    if (t == T_ - 1) nh[idx] = raw;
  } else {
    hpH[i] = (_Float16)0.f; hpL[i] = (_Float16)0.f;
  }
}

__global__ void logsoftmax_k(const float* __restrict__ logit, float* __restrict__ out)
{
  int b = threadIdx.x;                       // 64 threads
  if (b >= B_) return;
  float mx = -1e30f;
  for (int j = 0; j < NC_; ++j) mx = fmaxf(mx, logit[b * NC_ + j]);
  float sum = 0.f;
  for (int j = 0; j < NC_; ++j) sum += expf(logit[b * NC_ + j] - mx);
  float lse = mx + logf(sum);
  for (int j = 0; j < NC_; ++j) out[b * NC_ + j] = logit[b * NC_ + j] - lse;
}

// ---------------------------------------------------------------------------
extern "C" void kernel_launch(void* const* d_in, const int* in_sizes, int n_in,
                              void* d_out, int out_size, void* d_ws, size_t ws_size,
                              hipStream_t stream)
{
  (void)in_sizes; (void)n_in; (void)out_size; (void)ws_size;

  const int*   tokens = (const int*)  d_in[0];
  const float* masks  = (const float*)d_in[1];
  const int*   pos    = (const int*)  d_in[2];
  const int*   ner    = (const int*)  d_in[3];
  const float* hidden = (const float*)d_in[4];
  const float* encW   = (const float*)d_in[5];
  const float* nerW   = (const float*)d_in[6];
  const float* posW   = (const float*)d_in[7];
  const float* aggW   = (const float*)d_in[8];
  const float* aggB   = (const float*)d_in[9];
  const float* W0     = (const float*)d_in[10];
  const float* Ws     = (const float*)d_in[11];
  const float* decW   = (const float*)d_in[12];
  const float* decB   = (const float*)d_in[13];

  // ---- bump allocator over d_ws (total ~95 MB) ----
  char* base = (char*)d_ws;
  size_t off = 0;
  auto alloc = [&](size_t bytes) -> void* {
    void* p = base + off;
    off += (bytes + 255) & ~(size_t)255;
    return p;
  };

  _Float16* WsTh = (_Float16*)alloc((size_t)8 * D2_ * KC * 2);
  _Float16* WsTl = (_Float16*)alloc((size_t)8 * D2_ * KC * 2);
  _Float16* W0Th = (_Float16*)alloc((size_t)D2_ * KW0 * 2);
  _Float16* W0Tl = (_Float16*)alloc((size_t)D2_ * KW0 * 2);
  _Float16* aggTh = (_Float16*)alloc((size_t)D_ * KAG * 2);
  _Float16* aggTl = (_Float16*)alloc((size_t)D_ * KAG * 2);
  _Float16* decTh = (_Float16*)alloc((size_t)NC_ * KC * 2);
  _Float16* decTl = (_Float16*)alloc((size_t)NC_ * KC * 2);
  _Float16* Xh    = (_Float16*)alloc((size_t)T_ * B_ * KC * 2);
  _Float16* Xl    = (_Float16*)alloc((size_t)T_ * B_ * KC * 2);
  _Float16* stH   = (_Float16*)alloc((size_t)9 * B_ * KC * 2);
  _Float16* stL   = (_Float16*)alloc((size_t)9 * B_ * KC * 2);
  _Float16* hpH   = (_Float16*)alloc((size_t)B_ * KC * 2);
  _Float16* hpL   = (_Float16*)alloc((size_t)B_ * KC * 2);
  _Float16* acH   = (_Float16*)alloc((size_t)B_ * KC * 2);
  _Float16* acL   = (_Float16*)alloc((size_t)B_ * KC * 2);
  float* Sst   = (float*)alloc((size_t)9 * BD * 4);
  float* hprev = (float*)alloc((size_t)BD * 4);
  float* accum = (float*)alloc((size_t)BD * 4);
  float* logit = (float*)alloc((size_t)B_ * NC_ * 4);

  // combined f16 pair (8192*928 halves = 15.2MB each) overlays the WsT region
  // (23.5MB each); WsT is converted only AFTER agg_gemm consumed combined.
  _Float16* combH = WsTh;
  _Float16* combL = WsTl;

  float* out = (float*)d_out;      // [0,2688) log_prob, then new_hidden
  float* nh  = out + B_ * NC_;

  // ---- preprocessing ----
  zero_f32<<<(BD + 255) / 256, 256, 0, stream>>>(accum, BD);
  build_combined<<<(8192 * KAG + 255) / 256, 256, 0, stream>>>(
      tokens, ner, pos, encW, nerW, posW, combH, combL);
  conv_aggT<<<(D_ * KAG + 255) / 256, 256, 0, stream>>>(aggW, aggTh, aggTl);
  dim3 agrid(54, 128);
  agg_gemm<<<agrid, 128, 0, stream>>>(combH, combL, aggTh, aggTl, aggB, Xh, Xl);

  (void)hipMemcpyAsync(hprev, hidden, (size_t)BD * sizeof(float),
                       hipMemcpyDeviceToDevice, stream);
  conv_pair64<<<(B_ * KC + 255) / 256, 256, 0, stream>>>(hidden, hpH, hpL);

  // weight conversions (WsT overwrites combined region - after agg_gemm)
  conv_W0T<<<(D2_ * KW0 + 255) / 256, 256, 0, stream>>>(W0, W0Th, W0Tl);
  conv_WsT<<<(8 * D2_ * KC + 255) / 256, 256, 0, stream>>>(Ws, WsTh, WsTl);
  conv_decT<<<(NC_ * KC + 255) / 256, 256, 0, stream>>>(decW, decTh, decTl);

  // dependency-levelized step groups (step i reads state preds[i]):
  //   {0} -> {1,2,3} -> {4,6} -> {5,7}
  const int g1 = 0;
  const int g2 = 1 | (2 << 8) | (3 << 16);
  const int g3 = 4 | (6 << 8);
  const int g4 = 5 | (7 << 8);

  // ---- recurrent scan ----
  for (int t = 0; t < T_; ++t) {
    gemm0<<<54, 128, 0, stream>>>(
        Xh + (size_t)t * B_ * KC, Xl + (size_t)t * B_ * KC,
        hpH, hpL, W0Th, W0Tl, hprev, Sst, stH, stL);
    step_gemm<<<dim3(54, 1), 128, 0, stream>>>(stH, stL, Sst, WsTh, WsTl, g1);
    step_gemm<<<dim3(54, 3), 128, 0, stream>>>(stH, stL, Sst, WsTh, WsTl, g2);
    step_gemm<<<dim3(54, 2), 128, 0, stream>>>(stH, stL, Sst, WsTh, WsTl, g3);
    step_gemm<<<dim3(54, 2), 128, 0, stream>>>(stH, stL, Sst, WsTh, WsTl, g4);
    cell_reduce<<<(B_ * KC + 255) / 256, 256, 0, stream>>>(
        Sst, hprev, hpH, hpL, accum, masks, nh, t);
  }

  // ---- decoder + log_softmax ----
  conv_pair64<<<(B_ * KC + 255) / 256, 256, 0, stream>>>(accum, acH, acL);
  dec_gemm<<<3, 128, 0, stream>>>(acH, acL, decTh, decTl, decB, logit);
  logsoftmax_k<<<1, 64, 0, stream>>>(logit, out);
}
